// APPNP_41807211659448
// MI455X (gfx1250) — matrix operand order repaired and checked by reading, not compiled, not run on hardware
//
#include <hip/hip_runtime.h>
#include <hip/hip_bf16.h>

typedef __attribute__((ext_vector_type(16))) __bf16 v16bf;
typedef __attribute__((ext_vector_type(8)))  float  v8f;

#define NNODES 100000
#define ALPHA  0.1f
#define KHOPS  10

// ---------------------------------------------------------------------------
// Weight transpose + fp32 -> bf16 convert:  W[K][C] -> WT[C][K]
// ---------------------------------------------------------------------------
__global__ void convert_wT_kernel(const float* __restrict__ W,
                                  __bf16* __restrict__ WT, int K, int C) {
    int t = blockIdx.x * blockDim.x + threadIdx.x;
    if (t >= K * C) return;
    int k = t / C, c = t % C;
    WT[(size_t)c * K + k] = (__bf16)W[t];
}

// ---------------------------------------------------------------------------
// WMMA GEMM: Out[N x outC] = act(A[N x K] * WT^T + bias)
//
// Block tile: 32 rows x (WAVES/2 * NT * 16) cols.
//   - A panel (32 x K) staged once in LDS (fp32->bf16 converted on the way in),
//     row-padded by 8 bf16 so the 16 fragment rows hit all 64 LDS banks.
//   - Each wave owns 1 row tile x NT col tiles: per 32-K step it reads one A
//     fragment from LDS (2x ds_load_b128) and issues NT wmma ops, reusing A.
//   - B (weights^T, bf16) is L2-resident; 2x global_load_b128 per tile.
// Fragment layouts per CDNA5 ISA 7.12.2 (16-bit A 16x32, B 32x16).
// ---------------------------------------------------------------------------
template<bool A_IS_F32, bool RELU, bool OUT_BF16, int K, int NT>
__global__ void __launch_bounds__(256)
wmma_mlp_gemm(const void* __restrict__ Aptr,
              const __bf16* __restrict__ WT,
              const float* __restrict__ bias,
              void* __restrict__ Outptr,
              int outC)
{
    constexpr int ROWS = 32;
    constexpr int PAD  = 8;              // bf16 pad per row -> 4-bank shift/row
    constexpr int LDK  = K + PAD;
    __shared__ __bf16 smem[ROWS * LDK];

    const int tid    = threadIdx.x;
    const int lane   = tid & 31;
    const int wave   = tid >> 5;
    const int laneHi = lane >> 4;        // 0: lanes 0-15, 1: lanes 16-31
    const int lane16 = lane & 15;
    const int m0     = blockIdx.x * ROWS;

    // ---- cooperative stage of A panel into LDS (convert fp32->bf16 if needed)
    if (A_IS_F32) {
        const float* A = (const float*)Aptr;
        for (int i = tid; i < ROWS * K / 4; i += blockDim.x) {
            int r  = i / (K / 4);
            int k4 = (i % (K / 4)) * 4;
            float4 f = *(const float4*)(A + (size_t)(m0 + r) * K + k4);
            union { __bf16 h[4]; uint2 u; } t;
            t.h[0] = (__bf16)f.x; t.h[1] = (__bf16)f.y;
            t.h[2] = (__bf16)f.z; t.h[3] = (__bf16)f.w;
            *(uint2*)(smem + r * LDK + k4) = t.u;
        }
    } else {
        const __bf16* A = (const __bf16*)Aptr;
        for (int i = tid; i < ROWS * K / 8; i += blockDim.x) {
            int r  = i / (K / 8);
            int k8 = (i % (K / 8)) * 8;
            *(uint4*)(smem + r * LDK + k8) =
                *(const uint4*)(A + (size_t)(m0 + r) * K + k8);
        }
    }
    __syncthreads();

    // ---- wave -> (row tile, col group) mapping: 2 row tiles per block
    const int waves      = blockDim.x >> 5;
    const int wavesPerRT = waves >> 1;
    const int rowTile    = wave / wavesPerRT;          // 0 or 1
    const int cg         = wave % wavesPerRT;          // column group

    const __bf16* arow_lds = smem + (rowTile * 16 + lane16) * LDK;

    const __bf16* brow[NT];
    #pragma unroll
    for (int j = 0; j < NT; ++j)
        brow[j] = WT + (size_t)((cg * NT + j) * 16 + lane16) * K;

    v8f acc[NT] = {};

    #pragma unroll
    for (int kk = 0; kk < K; kk += 32) {
        // A fragment: halves 0..7 -> k=ka..ka+7, halves 8..15 -> k=ka+16..ka+23
        const int ka = kk + (laneHi ? 8 : 0);
        union { v16bf v; uint4 u[2]; } ua;
        ua.u[0] = *(const uint4*)(arow_lds + ka);
        ua.u[1] = *(const uint4*)(arow_lds + ka + 16);

        // B fragment: lane = column; halves 0..7 -> k=kb..kb+7, halves 8..15 -> k=kb+16..kb+23
        const int kb = kk + (laneHi ? 8 : 0);
        #pragma unroll
        for (int j = 0; j < NT; ++j) {
            union { v16bf v; uint4 u[2]; } ub;
            ub.u[0] = *(const uint4*)(brow[j] + kb);
            ub.u[1] = *(const uint4*)(brow[j] + kb + 16);
            acc[j] = __builtin_amdgcn_wmma_f32_16x16x32_bf16(
                         false, ua.v, false, ub.v,
                         (short)0, acc[j], false, false);
        }
    }

    // ---- epilogue: bias (+ReLU) and store
    #pragma unroll
    for (int j = 0; j < NT; ++j) {
        const int n  = (cg * NT + j) * 16 + lane16;
        const float bv = bias[n];
        #pragma unroll
        for (int r = 0; r < 8; ++r) {
            const int m = m0 + rowTile * 16 + r + (laneHi ? 8 : 0);
            float v = acc[j][r] + bv;
            if (RELU) v = fmaxf(v, 0.0f);
            if (OUT_BF16) ((__bf16*)Outptr)[(size_t)m * outC + n] = (__bf16)v;
            else          ((float *)Outptr)[(size_t)m * outC + n] = v;
        }
    }
}

// ---------------------------------------------------------------------------
// Graph propagation kernels (working set ~45 MB -> L2-resident atomics)
// ---------------------------------------------------------------------------
__global__ void degree_kernel(const int* __restrict__ esrc, const int* __restrict__ edst,
                              float* __restrict__ dout, float* __restrict__ din, int nedges) {
    int e = blockIdx.x * blockDim.x + threadIdx.x;
    if (e >= nedges) return;
    atomicAdd(&dout[esrc[e]], 1.0f);
    atomicAdd(&din [edst[e]], 1.0f);
}

__global__ void norm_kernel(const float* __restrict__ dout, const float* __restrict__ din,
                            float* __restrict__ nsrc, float* __restrict__ ndst, int n) {
    int i = blockIdx.x * blockDim.x + threadIdx.x;
    if (i >= n) return;
    nsrc[i] = rsqrtf(fmaxf(dout[i], 1.0f));
    ndst[i] = rsqrtf(fmaxf(din [i], 1.0f));
}

__global__ void scale_kernel(const float* __restrict__ h, const float* __restrict__ nsrc,
                             float* __restrict__ hs, int total) {
    int t = blockIdx.x * blockDim.x + threadIdx.x;
    if (t >= total) return;
    hs[t] = h[t] * nsrc[t >> 4];
}

// 4 lanes per edge, 4 channels each: float4 gather + 4 f32 atomic scatter-adds
__global__ void scatter_kernel(const float* __restrict__ hs,
                               const int* __restrict__ esrc, const int* __restrict__ edst,
                               float* __restrict__ hnext, int nedges) {
    int t = blockIdx.x * blockDim.x + threadIdx.x;
    int e  = t >> 2;
    int c4 = (t & 3) << 2;
    if (e >= nedges) return;
    __builtin_prefetch(&esrc[e + 2048], 0, 1);   // global_prefetch_b8 on edge stream
    __builtin_prefetch(&edst[e + 2048], 0, 1);
    int s = esrc[e], d = edst[e];
    float4 m = *(const float4*)(hs + (size_t)s * 16 + c4);
    float* o = hnext + (size_t)d * 16 + c4;
    atomicAdd(o + 0, m.x);
    atomicAdd(o + 1, m.y);
    atomicAdd(o + 2, m.z);
    atomicAdd(o + 3, m.w);
}

// h = (1-a)*agg*ndst + a*h0 ; also pre-scale hs = h*nsrc for the next hop
__global__ void combine_kernel(const float* __restrict__ hnext, const float* __restrict__ h0,
                               const float* __restrict__ ndst, const float* __restrict__ nsrc,
                               float* __restrict__ hout, float* __restrict__ hs,
                               int total, int writeHs) {
    int t = blockIdx.x * blockDim.x + threadIdx.x;
    if (t >= total) return;
    int node = t >> 4;
    float agg = hnext[t] * ndst[node];
    float h = (1.0f - ALPHA) * agg + ALPHA * h0[t];
    hout[t] = h;
    if (writeHs) hs[t] = h * nsrc[node];
}

// ---------------------------------------------------------------------------
extern "C" void kernel_launch(void* const* d_in, const int* in_sizes, int n_in,
                              void* d_out, int out_size, void* d_ws, size_t ws_size,
                              hipStream_t stream) {
    const float* features = (const float*)d_in[0];
    const int*   esrc     = (const int*)  d_in[1];
    const int*   edst     = (const int*)  d_in[2];
    const float* W0 = (const float*)d_in[3];
    const float* b0 = (const float*)d_in[4];
    const float* W1 = (const float*)d_in[5];
    const float* b1 = (const float*)d_in[6];
    const float* W2 = (const float*)d_in[7];
    const float* b2 = (const float*)d_in[8];

    const int N      = NNODES;
    const int nedges = in_sizes[1];
    const int inF = 512, hid = 256, cls = 16;

    // ---- workspace carve-up (256B aligned) ----
    char* ws = (char*)d_ws;
    size_t off = 0;
    auto alloc = [&](size_t bytes) -> char* {
        char* p = ws + off;
        off = (off + bytes + 255) & ~(size_t)255;
        return p;
    };
    __bf16* W0T  = (__bf16*)alloc((size_t)hid * inF * 2);
    __bf16* W1T  = (__bf16*)alloc((size_t)hid * hid * 2);
    __bf16* W2T  = (__bf16*)alloc((size_t)cls * hid * 2);
    __bf16* h1   = (__bf16*)alloc((size_t)N * hid * 2);
    __bf16* h2   = (__bf16*)alloc((size_t)N * hid * 2);
    float*  h0   = (float*) alloc((size_t)N * cls * 4);
    float*  hcur = (float*) alloc((size_t)N * cls * 4);
    float*  hnxt = (float*) alloc((size_t)N * cls * 4);
    float*  hs   = (float*) alloc((size_t)N * cls * 4);
    float*  dego = (float*) alloc((size_t)N * 4);
    float*  degi = (float*) alloc((size_t)N * 4);
    float*  nsrc = (float*) alloc((size_t)N * 4);
    float*  ndst = (float*) alloc((size_t)N * 4);

    // ---- weight convert/transpose to bf16 ----
    {
        int t0 = inF * hid, t1 = hid * hid, t2 = hid * cls;
        convert_wT_kernel<<<(t0 + 255) / 256, 256, 0, stream>>>(W0, W0T, inF, hid);
        convert_wT_kernel<<<(t1 + 255) / 256, 256, 0, stream>>>(W1, W1T, hid, hid);
        convert_wT_kernel<<<(t2 + 255) / 256, 256, 0, stream>>>(W2, W2T, hid, cls);
    }

    // ---- MLP encoder via WMMA (LDS-staged A panels) ----
    dim3 gA(N / 32, 1);   // 8 waves: 2 row tiles x 4 col groups x 4 col tiles
    wmma_mlp_gemm<true,  true,  true,  512, 4><<<gA, 256, 0, stream>>>(features, W0T, b0, h1, hid);
    wmma_mlp_gemm<false, true,  true,  256, 4><<<gA, 256, 0, stream>>>(h1,       W1T, b1, h2, hid);
    // 16 output classes: 2 waves = 2 row tiles x 1 col tile
    wmma_mlp_gemm<false, false, false, 256, 1><<<gA, 64,  0, stream>>>(h2,       W2T, b2, h0, cls);

    // ---- degree + symmetric normalization ----
    hipMemsetAsync(dego, 0, (size_t)N * 4, stream);
    hipMemsetAsync(degi, 0, (size_t)N * 4, stream);
    degree_kernel<<<(nedges + 255) / 256, 256, 0, stream>>>(esrc, edst, dego, degi, nedges);
    norm_kernel<<<(N + 255) / 256, 256, 0, stream>>>(dego, degi, nsrc, ndst, N);

    // ---- K hops of APPNP ----
    const int total = N * cls;
    scale_kernel<<<(total + 255) / 256, 256, 0, stream>>>(h0, nsrc, hs, total);
    for (int hop = 0; hop < KHOPS; ++hop) {
        hipMemsetAsync(hnxt, 0, (size_t)total * 4, stream);
        long st = (long)nedges * 4;
        scatter_kernel<<<(st + 255) / 256, 256, 0, stream>>>(hs, esrc, edst, hnxt, nedges);
        const int last = (hop == KHOPS - 1);
        combine_kernel<<<(total + 255) / 256, 256, 0, stream>>>(
            hnxt, h0, ndst, nsrc, last ? (float*)d_out : hcur, hs, total, last ? 0 : 1);
    }
}
